// JukeboxAutoEncoder_1391569404582
// MI455X (gfx1250) — matrix operand order repaired and checked by reading, compiled, not run on hardware
//
#include <hip/hip_runtime.h>
#include <hip/hip_bf16.h>

#define FCH   64
#define BATCH 4
#define NEMB  512
#define TTILE 64
#define ACT_ROWS 132

typedef __attribute__((ext_vector_type(16))) __bf16 v16bf;
typedef __attribute__((ext_vector_type(8)))  float  v8f;

union ABf { v16bf v; unsigned int u[8]; };
union CDf { v8f  v; float f[8]; };

__device__ inline unsigned short f2bf(float f) {
  union { float f; unsigned int u; } x; x.f = f;
  unsigned int r = x.u + 0x7FFFu + ((x.u >> 16) & 1u);   // round-to-nearest-even
  return (unsigned short)(r >> 16);
}

// ---------------------------------------------------------------------------
// Generic WMMA conv: y[b, i*out_stride+out_off, co] =
//   relu( sum_tap sum_ci x[b, i*IN_STRIDE + toff[tap], ci] * w[wsel[tap], ci, co] + bias[co] ) (+ res)
// One block = 64 output positions (i) x all 64 channels for one batch.
// NTAPS / IN_STRIDE compile-time -> fully unrolled straight-line WMMA chain.
// ---------------------------------------------------------------------------
template <int NTAPS, int IN_STRIDE>
__global__ __launch_bounds__(256) void k_conv_gemm(
    const float* __restrict__ in, const float* __restrict__ w,
    const float* __restrict__ bias, const float* __restrict__ res,
    float* __restrict__ out,
    int T_in, int T_out, int n_i,
    int a0, int a1, int a2, int a3,
    int s0, int s1, int s2, int s3,
    int out_stride, int out_off, int do_relu)
{
  __shared__ unsigned short sA[ACT_ROWS * FCH];   // activations, bf16, [row][ci]
  __shared__ unsigned short sW[4 * FCH * FCH];    // weights, bf16, [tap][co][ci] (transposed)

  const int tid = threadIdx.x;
  const int b   = blockIdx.y;
  const int i0  = blockIdx.x * TTILE;

  const int ta[4] = {a0, a1, a2, a3};
  const int wk[4] = {s0, s1, s2, s3};
  const int span  = ta[NTAPS - 1] - ta[0];
  const int rows  = (TTILE - 1) * IN_STRIDE + span + 1;
  const int baseT = i0 * IN_STRIDE + ta[0];

  for (int idx = tid; idx < rows * FCH; idx += 256) {
    int r = idx >> 6, c = idx & 63;
    int tg = baseT + r;
    float v = (tg >= 0 && tg < T_in) ? in[((size_t)b * T_in + tg) * FCH + c] : 0.0f;
    sA[idx] = f2bf(v);
  }
  for (int idx = tid; idx < NTAPS * FCH * FCH; idx += 256) {
    int tap = idx >> 12;
    int rem = idx & 4095;
    int ci = rem >> 6, co = rem & 63;
    sW[(tap * FCH + co) * FCH + ci] = f2bf(w[((size_t)wk[tap] * FCH + ci) * FCH + co]);
  }
  __syncthreads();

  const int wave = tid >> 5, lane = tid & 31;
  const int g = lane >> 4, ln = lane & 15;

  for (int tile = wave; tile < 16; tile += 8) {
    const int tm = tile >> 2, tn = tile & 3;
    const int co = tn * 16 + ln;
    CDf acc;
#pragma unroll
    for (int j = 0; j < 8; ++j) acc.f[j] = 0.0f;

#pragma unroll
    for (int tap = 0; tap < NTAPS; ++tap) {
      const int roff = ta[tap] - ta[0];
      const int rA = (tm * 16 + ln) * IN_STRIDE + roff;     // A: lane holds row M=ln
#pragma unroll
      for (int chunk = 0; chunk < 2; ++chunk) {
        ABf a, bm;
        const unsigned short* arow = &sA[rA * FCH + chunk * 32];
#pragma unroll
        for (int j = 0; j < 8; ++j) {
          int kp = ((j >> 2) << 4) + (g << 3) + ((j & 3) << 1);  // A K-pair layout
          a.u[j] = *(const unsigned int*)(arow + kp);
        }
        const unsigned short* brow = &sW[(tap * FCH + co) * FCH + chunk * 32];
#pragma unroll
        for (int j = 0; j < 8; ++j) {
          int kp = ((j >> 2) << 4) + (g << 3) + ((j & 3) << 1);  // B K-pair layout
          bm.u[j] = *(const unsigned int*)(brow + kp);
        }
        acc.v = __builtin_amdgcn_wmma_f32_16x16x32_bf16(
            false, a.v, false, bm.v, (short)0, acc.v, false, false);
      }
    }

    const float bsv = bias[co];
#pragma unroll
    for (int j = 0; j < 8; ++j) {
      int m  = g * 8 + j;                                   // C/D: VGPR j -> row
      int ig = i0 + tm * 16 + m;
      if (ig >= n_i) continue;
      float v = acc.f[j] + bsv;
      if (do_relu) v = fmaxf(v, 0.0f);
      size_t o = ((size_t)b * T_out + (size_t)ig * out_stride + out_off) * FCH + co;
      if (res) v += res[o];
      out[o] = v;
    }
  }
}

// ---------------------------------------------------------------------------
// Vector quantiser: dot via WMMA, dist/argmin/similarity/z_q/loss
// ---------------------------------------------------------------------------
__global__ __launch_bounds__(256) void k_vq(
    const float* __restrict__ ze, const unsigned short* __restrict__ cb_bf,
    const float* __restrict__ cb_f, const float* __restrict__ ne,
    float* __restrict__ zq, float* __restrict__ sim,
    float* __restrict__ ids_out, float* __restrict__ loss, int BT)
{
  __shared__ unsigned short sZ[16 * FCH];
  __shared__ float sDist[16 * NEMB];
  __shared__ float sNz[16];
  __shared__ float sPv[256];
  __shared__ int   sPi[256];
  __shared__ int   sId[16];

  const int tid = threadIdx.x;
  const int r0  = blockIdx.x * 16;

  for (int idx = tid; idx < 16 * FCH; idx += 256) {
    int row = idx >> 6, c = idx & 63;
    sZ[idx] = f2bf(ze[((size_t)(r0 + row)) * FCH + c]);
  }
  if (tid < 16) {
    float s = 0.f;
    const float* zr = ze + ((size_t)(r0 + tid)) * FCH;
    for (int c = 0; c < FCH; ++c) { float v = zr[c]; s += v * v; }
    sNz[tid] = s;
  }
  __syncthreads();

  const int wave = tid >> 5, lane = tid & 31;
  const int g = lane >> 4, ln = lane & 15;

#pragma unroll
  for (int tt = 0; tt < 4; ++tt) {
    const int tnb = wave * 4 + tt;                 // 32 N-tiles of 16 codes
    const int n = tnb * 16 + ln;
    CDf acc;
#pragma unroll
    for (int j = 0; j < 8; ++j) acc.f[j] = 0.0f;
#pragma unroll
    for (int chunk = 0; chunk < 2; ++chunk) {
      ABf a, bm;
      const unsigned short* arow = &sZ[ln * FCH + chunk * 32];
#pragma unroll
      for (int j = 0; j < 8; ++j) {
        int kp = ((j >> 2) << 4) + (g << 3) + ((j & 3) << 1);
        a.u[j] = *(const unsigned int*)(arow + kp);
      }
      const unsigned short* brow = cb_bf + (size_t)n * FCH + chunk * 32;
#pragma unroll
      for (int j = 0; j < 8; ++j) {
        int kp = ((j >> 2) << 4) + (g << 3) + ((j & 3) << 1);
        bm.u[j] = *(const unsigned int*)(brow + kp);
      }
      acc.v = __builtin_amdgcn_wmma_f32_16x16x32_bf16(
          false, a.v, false, bm.v, (short)0, acc.v, false, false);
    }
    const float nev = ne[n];
    const float rne = rsqrtf(nev);
#pragma unroll
    for (int j = 0; j < 8; ++j) {
      int m = g * 8 + j;
      float dot = acc.f[j];
      sDist[m * NEMB + n] = nev + sNz[m] - 2.0f * dot;
      sim[((size_t)(r0 + m)) * NEMB + n] = dot * rsqrtf(sNz[m]) * rne;
    }
  }
  __syncthreads();

  {  // per-(row,segment) partial argmin over 32 codes
    int seg = tid & 15;
    const float* dr = &sDist[(tid >> 4) * NEMB + seg * 32];
    float bv = dr[0]; int bi = seg * 32;
    for (int k = 1; k < 32; ++k) {
      float v = dr[k];
      if (v < bv) { bv = v; bi = seg * 32 + k; }
    }
    sPv[tid] = bv; sPi[tid] = bi;
  }
  __syncthreads();
  if (tid < 16) {
    float bv = sPv[tid * 16]; int bi = sPi[tid * 16];
    for (int k = 1; k < 16; ++k) {
      float v = sPv[tid * 16 + k]; int i = sPi[tid * 16 + k];
      if (v < bv || (v == bv && i < bi)) { bv = v; bi = i; }
    }
    sId[tid] = bi;
    ids_out[r0 + tid] = (float)bi;
    atomicAdd(loss, sqrtf(fmaxf(bv, 0.0f)) * (1.25f / (float)BT)); // codebook + 0.25*commit
  }
  __syncthreads();
  for (int idx = tid; idx < 16 * FCH; idx += 256) {
    int row = idx >> 6, c = idx & 63;
    zq[((size_t)(r0 + row)) * FCH + c] = cb_f[(size_t)sId[row] * FCH + c];
  }
}

// ---------------------------------------------------------------------------
// Input conv: 1 -> 64 channels, K=4, stride 2, SAME (pad-left 1), ReLU
// ---------------------------------------------------------------------------
__global__ __launch_bounds__(256) void k_down0(
    const float* __restrict__ x, const float* __restrict__ w,
    const float* __restrict__ bias, float* __restrict__ out, int T_in, int T_out)
{
  int idx = blockIdx.x * 256 + threadIdx.x;
  if (idx >= BATCH * T_out * FCH) return;
  int co = idx & 63;
  int t  = (idx >> 6) % T_out;
  int b  = idx / (T_out * FCH);
  float s = bias[co];
#pragma unroll
  for (int k = 0; k < 4; ++k) {
    int tg = 2 * t - 1 + k;
    if (tg >= 0 && tg < T_in) s += x[(size_t)b * T_in + tg] * w[k * FCH + co];
  }
  out[idx] = fmaxf(s, 0.0f);
}

// ---------------------------------------------------------------------------
// Output projection: 64 -> 1 channels, K=3 dil=1, SAME, no activation
// ---------------------------------------------------------------------------
__global__ __launch_bounds__(256) void k_proj(
    const float* __restrict__ h, const float* __restrict__ w,
    const float* __restrict__ bias, float* __restrict__ out, int T)
{
  int idx = blockIdx.x * 256 + threadIdx.x;
  if (idx >= BATCH * T) return;
  int t = idx % T, b = idx / T;
  float s = bias[0];
  for (int k = 0; k < 3; ++k) {
    int tg = t - 1 + k;
    if (tg < 0 || tg >= T) continue;
    const float* hr = h + ((size_t)b * T + tg) * FCH;
    const float* wr = w + k * FCH;
    for (int c = 0; c < FCH; ++c) s += hr[c] * wr[c];
  }
  out[idx] = s;
}

__global__ void k_cb_prep(const float* __restrict__ cb, unsigned short* __restrict__ cb_bf) {
  int idx = blockIdx.x * 256 + threadIdx.x;
  if (idx < NEMB * FCH) cb_bf[idx] = f2bf(cb[idx]);
}
__global__ void k_cb_norm(const float* __restrict__ cb, float* __restrict__ ne,
                          float* __restrict__ loss) {
  int idx = blockIdx.x * 256 + threadIdx.x;
  if (idx == 0) *loss = 0.0f;                  // deterministic across replays
  if (idx < NEMB) {
    float s = 0.f;
    for (int c = 0; c < FCH; ++c) { float v = cb[(size_t)idx * FCH + c]; s += v * v; }
    ne[idx] = s;
  }
}

// ---------------------------------------------------------------------------
extern "C" void kernel_launch(void* const* d_in, const int* in_sizes, int n_in,
                              void* d_out, int out_size, void* d_ws, size_t ws_size,
                              hipStream_t stream)
{
  (void)in_sizes; (void)n_in; (void)out_size; (void)ws_size;
  const float* x       = (const float*)d_in[0];
  const float* w_down0 = (const float*)d_in[1];
  const float* b_down0 = (const float*)d_in[2];
  const float* w_down  = (const float*)d_in[3];
  const float* b_down  = (const float*)d_in[4];
  const float* w_res_e = (const float*)d_in[5];
  const float* b_res_e = (const float*)d_in[6];
  const float* cb      = (const float*)d_in[7];
  const float* w_res_d = (const float*)d_in[8];
  const float* b_res_d = (const float*)d_in[9];
  const float* w_up    = (const float*)d_in[10];
  const float* b_up    = (const float*)d_in[11];
  const float* w_proj  = (const float*)d_in[12];
  const float* b_proj  = (const float*)d_in[13];

  float* out   = (float*)d_out;
  float* x_hat = out;                 // 4*65536*1
  float* loss  = out + 262144;        // scalar
  float* ids   = out + 262145;        // 4*8192
  float* sim   = out + 294913;        // 4*8192*512

  char* ws = (char*)d_ws;
  float* bufA = (float*)ws;                                       // 64 MB
  float* bufB = (float*)(ws + (size_t)64 * 1024 * 1024);          // 64 MB
  unsigned short* cb_bf = (unsigned short*)(ws + (size_t)128 * 1024 * 1024);
  float* ne = (float*)(ws + (size_t)128 * 1024 * 1024 + 65536);

  // dilated / stride-1 residual conv (3 taps)
  auto conv3 = [&](const float* in, const float* w, const float* bs, const float* res, float* o,
                   int T, int dil, int relu) {
    dim3 grid((T + TTILE - 1) / TTILE, BATCH);
    k_conv_gemm<3, 1><<<grid, 256, 0, stream>>>(in, w, bs, res, o, T, T, T,
                                                -dil, 0, dil, 0, 0, 1, 2, 0, 1, 0, relu);
  };
  // stride-2 downsample conv (4 taps, SAME pad-left 1)
  auto conv_down = [&](const float* in, const float* w, const float* bs, float* o, int Tin) {
    dim3 grid((Tin / 2 + TTILE - 1) / TTILE, BATCH);
    k_conv_gemm<4, 2><<<grid, 256, 0, stream>>>(in, w, bs, nullptr, o, Tin, Tin / 2, Tin / 2,
                                                -1, 0, 1, 2, 0, 1, 2, 3, 1, 0, 1);
  };
  // transposed conv stride 2 K=4 SAME, one phase (2 taps), outputs o = 2i+phase
  auto conv_up = [&](const float* in, const float* w, const float* bs, float* o, int Tin,
                     int t0, int t1, int w0, int w1, int phase) {
    dim3 grid((Tin + TTILE - 1) / TTILE, BATCH);
    k_conv_gemm<2, 1><<<grid, 256, 0, stream>>>(in, w, bs, nullptr, o, Tin, 2 * Tin, Tin,
                                                t0, t1, 0, 0, w0, w1, 0, 0, 2, phase, 1);
  };
  auto wre = [&](int blk, int r, int j) { return w_res_e + (size_t)((blk*4 + r)*2 + j) * 3 * FCH * FCH; };
  auto bre = [&](int blk, int r, int j) { return b_res_e + (size_t)((blk*4 + r)*2 + j) * FCH; };
  auto wrd = [&](int blk, int r, int j) { return w_res_d + (size_t)((blk*4 + r)*2 + j) * 3 * FCH * FCH; };
  auto brd = [&](int blk, int r, int j) { return b_res_d + (size_t)((blk*4 + r)*2 + j) * FCH; };

  // codebook prep (also zeroes loss slot before VQ atomics)
  k_cb_prep<<<(NEMB * FCH + 255) / 256, 256, 0, stream>>>(cb, cb_bf);
  k_cb_norm<<<2, 256, 0, stream>>>(cb, ne, loss);

  const int T0 = 65536;
  int T = T0 / 2;

  // ---- encoder ----
  k_down0<<<(BATCH * T * FCH + 255) / 256, 256, 0, stream>>>(x, w_down0, b_down0, bufA, T0, T);
  float* h = bufA; float* t = bufB;
  for (int blk = 0; blk < 3; ++blk) {
    if (blk > 0) {
      conv_down(h, w_down + (size_t)(blk - 1) * 4 * FCH * FCH, b_down + (size_t)(blk - 1) * FCH, t, T);
      T /= 2; float* tmp = h; h = t; t = tmp;
    }
    for (int r = 0; r < 4; ++r) {
      conv3(h, wre(blk, r, 0), bre(blk, r, 0), nullptr, t, T, 3, 1);
      conv3(t, wre(blk, r, 1), bre(blk, r, 1), h,       h, T, 1, 1);
    }
  }
  // h == z_e (bufA), T == 8192

  // ---- vector quantiser ----
  k_vq<<<(BATCH * T) / 16, 256, 0, stream>>>(h, cb_bf, cb, ne, t, sim, ids, loss, BATCH * T);
  { float* tmp = h; h = t; t = tmp; }   // h = z_q

  // ---- decoder ----
  for (int blk = 0; blk < 3; ++blk) {
    for (int r = 0; r < 4; ++r) {
      conv3(h, wrd(blk, r, 0), brd(blk, r, 0), nullptr, t, T, 1, 1);
      conv3(t, wrd(blk, r, 1), brd(blk, r, 1), h,       h, T, 3, 1);
    }
    // transposed conv: even o=2i: w0 @ x[i-1] + w2 @ x[i];  odd o=2i+1: w1 @ x[i] + w3 @ x[i+1]
    const float* wu = w_up + (size_t)blk * 4 * FCH * FCH;
    const float* bu = b_up + (size_t)blk * FCH;
    conv_up(h, wu, bu, t, T, -1, 0, 0, 2, 0);
    conv_up(h, wu, bu, t, T,  0, 1, 1, 3, 1);
    T *= 2; float* tmp = h; h = t; t = tmp;
  }

  // ---- projection ----
  k_proj<<<(BATCH * T + 255) / 256, 256, 0, stream>>>(h, w_proj, b_proj, x_hat, T);
}
